// MLGRU_89799176225621
// MI455X (gfx1250) — matrix operand order repaired and checked by reading, not compiled, not run on hardware
//
#include <hip/hip_runtime.h>

#define DIM 2048
#define THRESH 0.33f

typedef _Float16 v16h __attribute__((ext_vector_type(16)));
typedef _Float16 v8h  __attribute__((ext_vector_type(8)));
typedef _Float16 v4h  __attribute__((ext_vector_type(4)));
typedef float    v8f  __attribute__((ext_vector_type(8)));
typedef float    v4f  __attribute__((ext_vector_type(4)));
typedef int      v4i  __attribute__((vector_size(16)));

union FragAB { v16h v; v8h h[2]; };

#define BM 128
#define BN 64
#define BK 32
#define A_PITCH 40   // f16 elems, 80B rows: 16B aligned, bank-spread
#define B_PITCH 72   // f16 elems, 144B rows: 16B aligned, bank-spread

#if __has_builtin(__builtin_amdgcn_global_load_async_to_lds_b128)
#define ASYNC_LDS 1
#else
#define ASYNC_LDS 0
#endif

// 16-byte global -> LDS copy. Async path bypasses VGPRs and is tracked by
// ASYNCcnt (CDNA5 GLOBAL_LOAD_ASYNC_TO_LDS_B128); fallback is load+ds_store.
__device__ __forceinline__ void cp_b128(void* gsrc, void* ldst) {
#if ASYNC_LDS
  __builtin_amdgcn_global_load_async_to_lds_b128(
      (__attribute__((address_space(1))) v4i*)gsrc,
      (__attribute__((address_space(3))) v4i*)ldst,
      /*imm offset=*/0, /*cpol=*/0);
#else
  *(v8h*)ldst = *(const v8h*)gsrc;
#endif
}

template <int N>
__device__ __forceinline__ void wait_async() {
#if ASYNC_LDS
#if __has_builtin(__builtin_amdgcn_s_wait_asynccnt)
  __builtin_amdgcn_s_wait_asynccnt(N);
#else
  asm volatile("s_wait_asynccnt %0" ::"n"(N) : "memory");
#endif
#endif
}

__device__ __forceinline__ v8f wmma_16x16x32(v16h a, v16h b, v8f c) {
  // D = A(16x32 f16) x B(32x16 f16) + C(16x16 f32)
  return __builtin_amdgcn_wmma_f32_16x16x32_f16(
      /*neg_a=*/false, a, /*neg_b=*/false, b,
      /*c_mod=*/(short)0, c, /*reuse_a=*/false, /*reuse_b=*/false);
}

__device__ __forceinline__ float sigmoid_(float x) {
  return 1.0f / (1.0f + __expf(-x));
}

// ---------------- prep: x (fp32) -> fp16 ----------------
__global__ void prep_x_f16(const float* __restrict__ x,
                           _Float16* __restrict__ x16, int n4) {
  int i = blockIdx.x * blockDim.x + threadIdx.x;
  if (i < n4) {
    v4f f = *(const v4f*)(x + 4 * (size_t)i);
    v4h h;
    h.x = (_Float16)f.x; h.y = (_Float16)f.y;
    h.z = (_Float16)f.z; h.w = (_Float16)f.w;
    *(v4h*)(x16 + 4 * (size_t)i) = h;
  }
}

// ---- prep: ternarize + transpose weights -> fp16, Wt[k][n] = tern(W[n][k]) ----
__global__ void prep_w_tern_t(const float* __restrict__ w0,
                              const float* __restrict__ w1,
                              const float* __restrict__ w2,
                              const float* __restrict__ w3,
                              _Float16* __restrict__ wt) {
  __shared__ float tile[32][33];
  const float* w = (blockIdx.z == 0) ? w0 :
                   (blockIdx.z == 1) ? w1 :
                   (blockIdx.z == 2) ? w2 : w3;
  _Float16* out = wt + (size_t)blockIdx.z * DIM * DIM;
  int k0 = blockIdx.x * 32, n0 = blockIdx.y * 32;
  int tx = threadIdx.x, ty = threadIdx.y;
  tile[ty][tx] = w[(size_t)(n0 + ty) * DIM + k0 + tx];   // coalesced along k
  __syncthreads();
  float v = tile[tx][ty];                                // = W[n0+tx][k0+ty]
  float t = (fabsf(v) < THRESH) ? 0.0f : (v > 0.0f ? 1.0f : -1.0f);
  out[(size_t)(k0 + ty) * DIM + n0 + tx] = (_Float16)t;  // coalesced along n
}

// ---------------- fused GEMM: f, c, g share A = x16 ----------------
// Block: 128(M) x 64(N), K-step 32; 8 wave32s as 4x2, each wave 32x32.
// Double-buffered LDS, async global->LDS staging overlapped with WMMA.
__global__ __launch_bounds__(256)
void gemm_fcg_fused(const _Float16* __restrict__ x16,
                    const _Float16* __restrict__ wtf,
                    const _Float16* __restrict__ wtc,
                    const _Float16* __restrict__ wtg,
                    const float* __restrict__ bf,
                    const float* __restrict__ bc,
                    const float* __restrict__ bg,
                    const float* __restrict__ h_prev,
                    float* __restrict__ out_h,
                    _Float16* __restrict__ gh16) {
  __shared__ _Float16 sA[2][BM][A_PITCH];
  __shared__ _Float16 sB[2][3][BK][B_PITCH];

  const int tid   = threadIdx.x;
  const int lane  = tid & 31;
  const int wave  = tid >> 5;
  const int waveM = wave >> 1;   // 0..3
  const int waveN = wave & 1;    // 0..1
  const int m_blk = blockIdx.y * BM;
  const int n_blk = blockIdx.x * BN;

  const int lhalf = lane >> 4;
  const int lcol  = lane & 15;

  // cooperative-staging indices (5 x b128 per thread per K-step)
  const int ar  = tid >> 2;          // A row 0..63 (two passes)
  const int ac  = (tid & 3) * 8;     // A col (f16), 16B chunks
  const int br  = tid >> 3;          // B row 0..31
  const int bc8 = (tid & 7) * 8;     // B col (f16), 16B chunks

  v8f accF[2][2] = {}, accC[2][2] = {}, accG[2][2] = {};

  auto stage = [&](int k0, int buf) {
#pragma unroll
    for (int it = 0; it < 2; ++it) {
      int r = ar + it * 64;
      cp_b128((void*)(x16 + (size_t)(m_blk + r) * DIM + k0 + ac),
              (void*)&sA[buf][r][ac]);
    }
    size_t off = (size_t)(k0 + br) * DIM + n_blk + bc8;
    cp_b128((void*)(wtf + off), (void*)&sB[buf][0][br][bc8]);
    cp_b128((void*)(wtc + off), (void*)&sB[buf][1][br][bc8]);
    cp_b128((void*)(wtg + off), (void*)&sB[buf][2][br][bc8]);
  };

  const int NK = DIM / BK;
  stage(0, 0);  // prologue: stage k=0 into buffer 0

  for (int kt = 0; kt < NK; ++kt) {
    const int cur = kt & 1;
    if (kt + 1 < NK) {
      stage((kt + 1) * BK, cur ^ 1);   // overlap next-stage DMA with compute
      wait_async<5>();                 // in-order: current stage has landed
    } else {
      wait_async<0>();
    }
    __syncthreads();

    // ---- A fragments (CDNA5 16-bit A 16x32 layout) ----
    FragAB a[2];
#pragma unroll
    for (int mt = 0; mt < 2; ++mt) {
      const _Float16* ap = &sA[cur][waveM * 32 + mt * 16 + lcol][8 * lhalf];
      a[mt].h[0] = *(const v8h*)ap;         // slots 0..7
      a[mt].h[1] = *(const v8h*)(ap + 16);  // slots 8..15
    }
    // ---- B fragments (lane = N column, slots = K) + WMMA ----
#pragma unroll
    for (int nt = 0; nt < 2; ++nt) {
      int nloc = waveN * 32 + nt * 16;
      FragAB b0, b1, b2;
#pragma unroll
      for (int s = 0; s < 8; ++s) {
        b0.v[s]     = sB[cur][0][8 * lhalf + s][nloc + lcol];
        b0.v[8 + s] = sB[cur][0][16 + 8 * lhalf + s][nloc + lcol];
        b1.v[s]     = sB[cur][1][8 * lhalf + s][nloc + lcol];
        b1.v[8 + s] = sB[cur][1][16 + 8 * lhalf + s][nloc + lcol];
        b2.v[s]     = sB[cur][2][8 * lhalf + s][nloc + lcol];
        b2.v[8 + s] = sB[cur][2][16 + 8 * lhalf + s][nloc + lcol];
      }
#pragma unroll
      for (int mt = 0; mt < 2; ++mt) {
        accF[mt][nt] = wmma_16x16x32(a[mt].v, b0.v, accF[mt][nt]);
        accC[mt][nt] = wmma_16x16x32(a[mt].v, b1.v, accC[mt][nt]);
        accG[mt][nt] = wmma_16x16x32(a[mt].v, b2.v, accG[mt][nt]);
      }
    }
    __syncthreads();  // all readers done before this buffer is re-staged
  }

  // ---- fused epilogue: sigmoid/silu gating; write h (fp32) and g*h (fp16) ----
#pragma unroll
  for (int mt = 0; mt < 2; ++mt) {
#pragma unroll
    for (int nt = 0; nt < 2; ++nt) {
      int Ng = n_blk + waveN * 32 + nt * 16 + lcol;
      float bfv = bf[Ng], bcv = bc[Ng], bgv = bg[Ng];
#pragma unroll
      for (int r = 0; r < 8; ++r) {
        int Mg = m_blk + waveM * 32 + mt * 16 + 8 * lhalf + r;
        size_t idx = (size_t)Mg * DIM + Ng;
        float f    = sigmoid_(accF[mt][nt][r] + bfv);
        float cpre = accC[mt][nt][r] + bcv;
        float cact = cpre * sigmoid_(cpre);           // silu
        float h    = f * h_prev[idx] + (1.0f - f) * cact;
        float g    = sigmoid_(accG[mt][nt][r] + bgv);
        out_h[idx] = h;
        gh16[idx]  = (_Float16)(g * h);
      }
    }
  }
}

// ---------------- output GEMM: o = (g*h) @ To^T + bo ----------------
__global__ __launch_bounds__(256)
void gemm_o(const _Float16* __restrict__ gh16,
            const _Float16* __restrict__ wto,
            const float* __restrict__ bo,
            float* __restrict__ out_o) {
  __shared__ _Float16 sA[2][BM][A_PITCH];
  __shared__ _Float16 sB[2][BK][B_PITCH];

  const int tid   = threadIdx.x;
  const int lane  = tid & 31;
  const int wave  = tid >> 5;
  const int waveM = wave >> 1;
  const int waveN = wave & 1;
  const int m_blk = blockIdx.y * BM;
  const int n_blk = blockIdx.x * BN;

  const int lhalf = lane >> 4;
  const int lcol  = lane & 15;

  const int ar  = tid >> 2;
  const int ac  = (tid & 3) * 8;
  const int br  = tid >> 3;
  const int bc8 = (tid & 7) * 8;

  v8f acc[2][2] = {};

  auto stage = [&](int k0, int buf) {
#pragma unroll
    for (int it = 0; it < 2; ++it) {
      int r = ar + it * 64;
      cp_b128((void*)(gh16 + (size_t)(m_blk + r) * DIM + k0 + ac),
              (void*)&sA[buf][r][ac]);
    }
    cp_b128((void*)(wto + (size_t)(k0 + br) * DIM + n_blk + bc8),
            (void*)&sB[buf][br][bc8]);
  };

  const int NK = DIM / BK;
  stage(0, 0);

  for (int kt = 0; kt < NK; ++kt) {
    const int cur = kt & 1;
    if (kt + 1 < NK) {
      stage((kt + 1) * BK, cur ^ 1);
      wait_async<3>();
    } else {
      wait_async<0>();
    }
    __syncthreads();

    FragAB a[2];
#pragma unroll
    for (int mt = 0; mt < 2; ++mt) {
      const _Float16* ap = &sA[cur][waveM * 32 + mt * 16 + lcol][8 * lhalf];
      a[mt].h[0] = *(const v8h*)ap;
      a[mt].h[1] = *(const v8h*)(ap + 16);
    }
#pragma unroll
    for (int nt = 0; nt < 2; ++nt) {
      int nloc = waveN * 32 + nt * 16;
      FragAB b;
#pragma unroll
      for (int s = 0; s < 8; ++s) {
        b.v[s]     = sB[cur][8 * lhalf + s][nloc + lcol];
        b.v[8 + s] = sB[cur][16 + 8 * lhalf + s][nloc + lcol];
      }
#pragma unroll
      for (int mt = 0; mt < 2; ++mt) {
        acc[mt][nt] = wmma_16x16x32(a[mt].v, b.v, acc[mt][nt]);
      }
    }
    __syncthreads();
  }

#pragma unroll
  for (int mt = 0; mt < 2; ++mt) {
#pragma unroll
    for (int nt = 0; nt < 2; ++nt) {
      int Ng = n_blk + waveN * 32 + nt * 16 + lcol;
      float bov = bo[Ng];
#pragma unroll
      for (int r = 0; r < 8; ++r) {
        int Mg = m_blk + waveM * 32 + mt * 16 + 8 * lhalf + r;
        out_o[(size_t)Mg * DIM + Ng] = acc[mt][nt][r] + bov;
      }
    }
  }
}

extern "C" void kernel_launch(void* const* d_in, const int* in_sizes, int n_in,
                              void* d_out, int out_size, void* d_ws, size_t ws_size,
                              hipStream_t stream) {
  const float* x      = (const float*)d_in[0];
  const float* h_prev = (const float*)d_in[1];
  const float* wf = (const float*)d_in[2];
  const float* bf = (const float*)d_in[3];
  const float* wc = (const float*)d_in[4];
  const float* bc = (const float*)d_in[5];
  const float* wg = (const float*)d_in[6];
  const float* bg = (const float*)d_in[7];
  const float* wo = (const float*)d_in[8];
  const float* bo = (const float*)d_in[9];

  const int D    = DIM;
  const int Mtot = in_sizes[0] / D;   // B*S = 16384

  float* o_out = (float*)d_out;
  float* h_out = (float*)d_out + (size_t)Mtot * D;

  char* ws = (char*)d_ws;
  _Float16* x16  = (_Float16*)ws;                                   // Mtot*D f16
  _Float16* wt   = (_Float16*)(ws + (size_t)Mtot * D * 2);          // 4*D*D f16 (f,c,g,o)
  _Float16* gh16 = (_Float16*)(ws + (size_t)Mtot * D * 2 + (size_t)4 * D * D * 2);

  {
    int n4 = (Mtot * D) / 4;
    prep_x_f16<<<(n4 + 255) / 256, 256, 0, stream>>>(x, x16, n4);
  }
  {
    dim3 b(32, 32), g(D / 32, D / 32, 4);
    prep_w_tern_t<<<g, b, 0, stream>>>(wf, wc, wg, wo, wt);
  }
  {
    dim3 g(D / BN, Mtot / BM);
    gemm_fcg_fused<<<g, 256, 0, stream>>>(x16,
                                          wt + 0 * (size_t)D * D,
                                          wt + 1 * (size_t)D * D,
                                          wt + 2 * (size_t)D * D,
                                          bf, bc, bg, h_prev, h_out, gh16);
  }
  {
    dim3 g(D / BN, Mtot / BM);
    gemm_o<<<g, 256, 0, stream>>>(gh16, wt + 3 * (size_t)D * D, bo, o_out);
  }
}
